// LeakyUnit_60885456388846
// MI455X (gfx1250) — matrix operand order repaired and checked by reading, not compiled, not run on hardware
//
#include <hip/hip_runtime.h>

typedef __attribute__((ext_vector_type(16))) _Float16 v16h;
typedef __attribute__((ext_vector_type(8)))  _Float16 v8h;
typedef __attribute__((ext_vector_type(8)))  float    v8f;

#define N_TASKS 3
#define BB 32
#define HH 32
#define TT 32
#define FF 256
#define CT 4                 // channels per block = waves per block (wave32)
#define NTHREADS (CT * 32)

// ---------------- WMMA helpers (CDNA5 wave32 layouts, 16x16x32 f16) ----------------

__device__ __forceinline__ v8f wmma16(v16h a, v16h b, v8f c) {
  // (neg_a, A, neg_b, B, c_mod, C, reuse_a, reuse_b)
  return __builtin_amdgcn_wmma_f32_16x16x32_f16(false, a, false, b, (short)0, c, false, false);
}

// A operand (16x32 f16) from row-major 32x32 LDS tile m[row][k]:
// lane<16: M=lane, K = 0..7 then 16..23 ; lane>=16: M=lane-16, K = 8..15 then 24..31
__device__ __forceinline__ v16h ldsA(const _Float16* m, int mt, int lane) {
  const int row = mt * 16 + (lane & 15);
  const int kb  = (lane & 16) ? 8 : 0;
  v8h lo = *(const v8h*)(m + row * 32 + kb);
  v8h hi = *(const v8h*)(m + row * 32 + kb + 16);
  return __builtin_shufflevector(lo, hi, 0, 1, 2, 3, 4, 5, 6, 7, 8, 9, 10, 11, 12, 13, 14, 15);
}

// B operand (32x16 f16) from LDS tile stored so that the contraction index is
// contiguous: m[col][k].  lane<16: N=col=lane, K = 0..7 then 16..23 ; lane>=16: K = 8..15 then 24..31
__device__ __forceinline__ v16h ldsB(const _Float16* m, int nt, int lane) {
  const int col = nt * 16 + (lane & 15);
  const int kb  = (lane & 16) ? 8 : 0;
  v8h lo = *(const v8h*)(m + col * 32 + kb);
  v8h hi = *(const v8h*)(m + col * 32 + kb + 16);
  return __builtin_shufflevector(lo, hi, 0, 1, 2, 3, 4, 5, 6, 7, 8, 9, 10, 11, 12, 13, 14, 15);
}

// Store D (16x16 f32) as f16 into row-major 32x32 LDS tile m[M][N].
// lane<16: N=lane, M=v ; lane>=16: N=lane-16, M=8+v
__device__ __forceinline__ void stD(_Float16* m, int mt, int nt, int lane, v8f d) {
  const int col  = nt * 16 + (lane & 15);
  const int row0 = mt * 16 + ((lane & 16) ? 8 : 0);
#pragma unroll
  for (int v = 0; v < 8; ++v) m[(row0 + v) * 32 + col] = (_Float16)d[v];
}

// Fast activations: hardware v_rcp_f32 instead of IEEE division sequences.
__device__ __forceinline__ v8f sigmoid8(v8f x) {
  v8f r;
#pragma unroll
  for (int v = 0; v < 8; ++v) r[v] = __builtin_amdgcn_rcpf(1.0f + __expf(-x[v]));
  return r;
}

__device__ __forceinline__ v8f tanh8(v8f x) {
  v8f r;
#pragma unroll
  for (int v = 0; v < 8; ++v) {
    float t = __expf(2.0f * x[v]);                      // tanh = 1 - 2/(e^{2x}+1)
    r[v] = 1.0f - 2.0f * __builtin_amdgcn_rcpf(t + 1.0f);
  }
  return r;
}

// ---------------- Global -> LDS staging (coalesced float4 along channels) ----------------

// x tile: dst[cc][h][k] = src[(h*TT + k)*FF + c0 + cc]
__device__ __forceinline__ void stage_x(const float* __restrict__ src,
                                        _Float16 (*dst)[32][32], int c0, int tid) {
#pragma unroll
  for (int it = 0; it < (HH * TT) / NTHREADS; ++it) {
    const int hk = it * NTHREADS + tid;
    const float4 v = *(const float4*)(src + (size_t)hk * FF + c0);
    const int h = hk >> 5, k = hk & 31;
    dst[0][h][k] = (_Float16)v.x;
    dst[1][h][k] = (_Float16)v.y;
    dst[2][h][k] = (_Float16)v.z;
    dst[3][h][k] = (_Float16)v.w;
  }
}

// W tile, transposed so contraction k is contiguous: dst[cc][t][k] = src[(k*TT + t)*FF + c0 + cc]
__device__ __forceinline__ void stage_wT(const float* __restrict__ src,
                                         _Float16 (*dst)[32][32], int c0, int tid) {
#pragma unroll
  for (int it = 0; it < (HH * TT) / NTHREADS; ++it) {
    const int kt = it * NTHREADS + tid;
    const float4 v = *(const float4*)(src + (size_t)kt * FF + c0);
    const int k = kt >> 5, t = kt & 31;
    dst[0][t][k] = (_Float16)v.x;
    dst[1][t][k] = (_Float16)v.y;
    dst[2][t][k] = (_Float16)v.z;
    dst[3][t][k] = (_Float16)v.w;
  }
}

// ---------------- Fused LeakyUnit kernel ----------------
// grid = (FF/CT, BB, N_TASKS), block = CT waves; wave cc handles channel c0+cc.
// For output task i, the two pairs are p = 2i, 2i+1 with j = the other two tasks.

__global__ __launch_bounds__(NTHREADS) void leaky_unit_fused(
    const float* __restrict__ x0, const float* __restrict__ x1, const float* __restrict__ x2,
    const float* __restrict__ Wr, const float* __restrict__ Wz,
    const float* __restrict__ Wh, const float* __restrict__ U,
    float* __restrict__ out) {
  __shared__ _Float16 sXi[CT][32][32];   // [cc][h][k]
  __shared__ _Float16 sXj[CT][32][32];   // [cc][h][k]
  __shared__ _Float16 sWr[CT][32][32];   // [cc][t][k]
  __shared__ _Float16 sWz[CT][32][32];   // [cc][t][k]
  __shared__ _Float16 sWh[CT][32][32];   // [cc][t][k]
  __shared__ _Float16 sU [CT][32][32];   // [cc][t][k]
  __shared__ _Float16 sS0[CT][32][32];   // scratch: r, then zeta   [cc][M][N]
  __shared__ _Float16 sS1[CT][32][32];   // scratch: new_map        [cc][M][N]

  const int tid  = threadIdx.x;
  const int lane = tid & 31;
  const int cc   = tid >> 5;
  const int c0   = blockIdx.x * CT;
  const int b    = blockIdx.y;
  const int i    = blockIdx.z;

  const float* xs[3] = {x0, x1, x2};
  const size_t slab = (size_t)HH * TT * FF;   // per-b x slab == per-p W slab

  stage_x(xs[i] + (size_t)b * slab, sXi, c0, tid);

  v8f acc[2][2];
#pragma unroll
  for (int mt = 0; mt < 2; ++mt)
#pragma unroll
    for (int nt = 0; nt < 2; ++nt) acc[mt][nt] = (v8f){};

  for (int jj = 0; jj < 2; ++jj) {
    const int p = 2 * i + jj;
    const int j = (jj == 0) ? ((i == 0) ? 1 : 0) : ((i == 2) ? 1 : 2);
    const size_t wo = (size_t)p * slab;

    __syncthreads();   // previous iteration's reads of sXj/W done before restage
    stage_x (xs[j] + (size_t)b * slab, sXj, c0, tid);
    stage_wT(Wr + wo, sWr, c0, tid);
    stage_wT(Wz + wo, sWz, c0, tid);
    stage_wT(Wh + wo, sWh, c0, tid);
    stage_wT(U  + wo, sU , c0, tid);
    __syncthreads();

    const _Float16* Xi = &sXi[cc][0][0];
    const _Float16* Xj = &sXj[cc][0][0];
    _Float16* S0 = &sS0[cc][0][0];
    _Float16* S1 = &sS1[cc][0][0];

    v16h a_xi[2], a_xj[2];
#pragma unroll
    for (int mt = 0; mt < 2; ++mt) {
      a_xi[mt] = ldsA(Xi, mt, lane);
      a_xj[mt] = ldsA(Xj, mt, lane);
    }

    // r = sigmoid(Xi@Wr + Xj@Wr) -> S0
#pragma unroll
    for (int nt = 0; nt < 2; ++nt) {
      const v16h bw = ldsB(&sWr[cc][0][0], nt, lane);
#pragma unroll
      for (int mt = 0; mt < 2; ++mt) {
        v8f d = (v8f){};
        d = wmma16(a_xj[mt], bw, d);
        d = wmma16(a_xi[mt], bw, d);
        stD(S0, mt, nt, lane, sigmoid8(d));
      }
    }

    // (r .* Xj) as A operand, packed-f16 multiply in registers
    v16h a_rxj[2];
#pragma unroll
    for (int mt = 0; mt < 2; ++mt) a_rxj[mt] = ldsA(S0, mt, lane) * a_xj[mt];

    // new_map = tanh(Xi@U + (r.*Xj)@Wh) -> S1
#pragma unroll
    for (int nt = 0; nt < 2; ++nt) {
      const v16h bu = ldsB(&sU[cc][0][0], nt, lane);
      const v16h bh = ldsB(&sWh[cc][0][0], nt, lane);
#pragma unroll
      for (int mt = 0; mt < 2; ++mt) {
        v8f d = (v8f){};
        d = wmma16(a_xi[mt], bu, d);
        d = wmma16(a_rxj[mt], bh, d);
        stD(S1, mt, nt, lane, tanh8(d));
      }
    }

    // zeta = sigmoid(Xi@Wz + Xj@Wz) -> S0 (r is dead)
#pragma unroll
    for (int nt = 0; nt < 2; ++nt) {
      const v16h bz = ldsB(&sWz[cc][0][0], nt, lane);
#pragma unroll
      for (int mt = 0; mt < 2; ++mt) {
        v8f d = (v8f){};
        d = wmma16(a_xj[mt], bz, d);
        d = wmma16(a_xi[mt], bz, d);
        stD(S0, mt, nt, lane, sigmoid8(d));
      }
    }

    // terms += Xi@zeta^T + nm@(1-zeta)^T
    // B[k,t] = zeta[t,k]: zeta stored [M][N] row-major -> ldsB reads row t, contiguous k.
    v16h a_nm[2];
#pragma unroll
    for (int mt = 0; mt < 2; ++mt) a_nm[mt] = ldsA(S1, mt, lane);
#pragma unroll
    for (int nt = 0; nt < 2; ++nt) {
      const v16h bZ = ldsB(S0, nt, lane);
      v16h bZ1;
#pragma unroll
      for (int e = 0; e < 16; ++e) bZ1[e] = (_Float16)1.0f - bZ[e];
#pragma unroll
      for (int mt = 0; mt < 2; ++mt) {
        acc[mt][nt] = wmma16(a_xi[mt], bZ, acc[mt][nt]);
        acc[mt][nt] = wmma16(a_nm[mt], bZ1, acc[mt][nt]);
      }
    }
  }

  // out[i,b,M,N,c] = acc / N_TASKS   (D layout -> strided global stores)
  const float scale = 1.0f / (float)N_TASKS;
  float* ob = out + ((size_t)i * BB + b) * slab + c0 + cc;
  const int col_l = lane & 15;
  const int roff  = (lane & 16) ? 8 : 0;
#pragma unroll
  for (int mt = 0; mt < 2; ++mt)
#pragma unroll
    for (int nt = 0; nt < 2; ++nt) {
      const int colg = nt * 16 + col_l;
#pragma unroll
      for (int v = 0; v < 8; ++v) {
        const int rowg = mt * 16 + roff + v;
        ob[(size_t)rowg * TT * FF + (size_t)colg * FF] = acc[mt][nt][v] * scale;
      }
    }
}

extern "C" void kernel_launch(void* const* d_in, const int* in_sizes, int n_in,
                              void* d_out, int out_size, void* d_ws, size_t ws_size,
                              hipStream_t stream) {
  const float* x0 = (const float*)d_in[0];
  const float* x1 = (const float*)d_in[1];
  const float* x2 = (const float*)d_in[2];
  const float* Wr = (const float*)d_in[3];
  const float* Wz = (const float*)d_in[4];
  const float* Wh = (const float*)d_in[5];
  const float* U  = (const float*)d_in[6];
  float* out = (float*)d_out;

  dim3 grid(FF / CT, BB, N_TASKS);
  leaky_unit_fused<<<grid, NTHREADS, 0, stream>>>(x0, x1, x2, Wr, Wz, Wh, U, out);

  (void)in_sizes; (void)n_in; (void)out_size; (void)d_ws; (void)ws_size;
}
